// ProductBlock_44203803410587
// MI455X (gfx1250) — matrix operand order repaired and checked by reading, not compiled, not run on hardware
//
#include <hip/hip_runtime.h>

typedef __attribute__((ext_vector_type(16))) __bf16 v16bf;
typedef __attribute__((ext_vector_type(8)))  float  v8f;
typedef __attribute__((ext_vector_type(4)))  float  v4f;

#define N_NODES   400000
#define IRREPS    256
#define NFEAT     128
#define NELEM     118
#define ROWTILES  (N_NODES / 16)   // 25000, exact

__global__ __launch_bounds__(256)
void fused_poly_wmma_gemm(const float* __restrict__ feats,   // [N][256]
                          const int*   __restrict__ attrs,   // [N]
                          const float* __restrict__ cw,      // [2][118][256]
                          const float* __restrict__ lin_w,   // [256][128]
                          const float* __restrict__ lin_b,   // [128]
                          float*       __restrict__ out)     // [N][128]
{
    // lin_w transposed to n-major bf16: B-fragment for (k-chunk, n-tile) is
    // two runs of 8 contiguous bf16 per lane. 128*256*2 = 64 KB LDS.
    __shared__ __bf16 lwT[NFEAT][IRREPS];

    const int tid = threadIdx.x;

    // One-time cooperative transpose + f32->bf16 convert of lin_w into LDS.
    for (int idx = tid; idx < IRREPS * NFEAT; idx += 256) {
        const int k = idx >> 7;     // 0..255 (row of lin_w)
        const int n = idx & 127;    // 0..127 (col of lin_w)
        lwT[n][k] = (__bf16)lin_w[idx];
    }
    __syncthreads();

    const int lane = tid & 31;
    const int wave = tid >> 5;
    const int m    = lane & 15;          // row-within-tile / N column
    const bool hi  = lane >= 16;

    // Bias depends only on (nt, m): load once, fold into accumulator init.
    float bias[8];
    #pragma unroll
    for (int nt = 0; nt < 8; ++nt) bias[nt] = lin_b[nt * 16 + m];

    const int waveStride = gridDim.x * 8;

    for (int tile = blockIdx.x * 8 + wave; tile < ROWTILES; tile += waveStride) {
        const int row  = tile * 16 + m;                 // node handled by this lane
        const int attr = attrs[row];
        const float* __restrict__ xrow  = feats + (size_t)row * IRREPS;
        const float* __restrict__ w0row = cw + (size_t)attr * IRREPS;
        const float* __restrict__ w1row = cw + (size_t)(NELEM + attr) * IRREPS;

        // Prefetch next tile's feature row into cache (global_prefetch_b8).
        if (tile + waveStride < ROWTILES) {
            __builtin_prefetch(feats + (size_t)(row + waveStride * 16) * IRREPS, 0, 1);
        }

        // 16x128 f32 accumulator tile, pre-loaded with the bias so the
        // epilogue is a pure store.
        v8f acc[8];
        #pragma unroll
        for (int nt = 0; nt < 8; ++nt) {
            const float bv = bias[nt];
            #pragma unroll
            for (int j = 0; j < 8; ++j) acc[nt][j] = bv;
        }

        for (int kc = 0; kc < 8; ++kc) {
            const int kb = kc * 32;
            // 16-bit A 16x32 lane layout: lanes 0-15 -> K {0..7,16..23},
            // lanes 16-31 -> K {8..15,24..31} (pairs per VGPR).
            const int k0 = kb + (hi ? 8 : 0);
            const int k1 = k0 + 16;

            // Streaming data: non-temporal loads (read-once, keep out of L2).
            const v4f xa0 = __builtin_nontemporal_load((const v4f*)(xrow + k0));
            const v4f xa1 = __builtin_nontemporal_load((const v4f*)(xrow + k0 + 4));
            const v4f xb0 = __builtin_nontemporal_load((const v4f*)(xrow + k1));
            const v4f xb1 = __builtin_nontemporal_load((const v4f*)(xrow + k1 + 4));
            // Gather tables: regular-temporal (L2-resident, 236 KB total).
            const v4f pa0 = *(const v4f*)(w0row + k0);
            const v4f pa1 = *(const v4f*)(w0row + k0 + 4);
            const v4f pb0 = *(const v4f*)(w0row + k1);
            const v4f pb1 = *(const v4f*)(w0row + k1 + 4);
            const v4f qa0 = *(const v4f*)(w1row + k0);
            const v4f qa1 = *(const v4f*)(w1row + k0 + 4);
            const v4f qb0 = *(const v4f*)(w1row + k1);
            const v4f qb1 = *(const v4f*)(w1row + k1 + 4);

            // basis = x*(w0 + w1*x), computed in f32, rounded to bf16 for WMMA A.
            v16bf a;
            #pragma unroll
            for (int i = 0; i < 4; ++i) {
                a[i]      = (__bf16)(xa0[i] * (pa0[i] + qa0[i] * xa0[i]));
                a[4 + i]  = (__bf16)(xa1[i] * (pa1[i] + qa1[i] * xa1[i]));
                a[8 + i]  = (__bf16)(xb0[i] * (pb0[i] + qb0[i] * xb0[i]));
                a[12 + i] = (__bf16)(xb1[i] * (pb1[i] + qb1[i] * xb1[i]));
            }

            // 16-bit B 32x16 lane layout (matches A): lanes 0-15 hold K {0..7,16..23},
            // lanes 16-31 hold K {8..15,24..31}, N = lane%16 -> two 8-bf16 runs in lwT.
            const int kgrp = kb + (hi ? 8 : 0);

            #pragma unroll
            for (int nt = 0; nt < 8; ++nt) {
                typedef __attribute__((ext_vector_type(8))) __bf16 v8bf;
                const v8bf b0 = *(const v8bf*)&lwT[nt * 16 + m][kgrp];
                const v8bf b1 = *(const v8bf*)&lwT[nt * 16 + m][kgrp + 16];
                const v16bf b = __builtin_shufflevector(b0, b1, 0,1,2,3,4,5,6,7,8,9,10,11,12,13,14,15);
                acc[nt] = __builtin_amdgcn_wmma_f32_16x16x32_bf16(
                    /*neg_a=*/false, a, /*neg_b=*/false, b,
                    /*c_mod=*/(short)0, acc[nt],
                    /*reuse_a=*/false, /*reuse_b=*/false);
            }
        }

        // Epilogue: C/D layout -> VGPR j is M=j (lanes 0-15) or M=j+8
        // (lanes 16-31), N = lane%16. One 64-bit base per tile; every
        // (nt, j) offset is a compile-time constant (max 4032 B, fits the
        // 24-bit instruction offset) so stores become
        // global_store_b32 ..., off offset:imm with TH=NT.
        float* __restrict__ obase =
            out + (size_t)tile * 16 * NFEAT + (hi ? 8 * NFEAT : 0) + m;
        #pragma unroll
        for (int nt = 0; nt < 8; ++nt) {
            #pragma unroll
            for (int j = 0; j < 8; ++j) {
                __builtin_nontemporal_store(acc[nt][j], obase + j * NFEAT + nt * 16);
            }
        }
    }
}

extern "C" void kernel_launch(void* const* d_in, const int* in_sizes, int n_in,
                              void* d_out, int out_size, void* d_ws, size_t ws_size,
                              hipStream_t stream) {
    const float* feats = (const float*)d_in[0];
    const int*   attrs = (const int*)  d_in[1];
    const float* cw    = (const float*)d_in[2];
    const float* lin_w = (const float*)d_in[3];
    const float* lin_b = (const float*)d_in[4];
    float*       out   = (float*)d_out;
    (void)in_sizes; (void)n_in; (void)out_size; (void)d_ws; (void)ws_size;

    // 1024 blocks x 8 waves = 8192 waves grid-striding over 25000 row tiles.
    fused_poly_wmma_gemm<<<1024, 256, 0, stream>>>(feats, attrs, cw, lin_w, lin_b, out);
}
